// SkewedRelAttnBlock_82523501625503
// MI455X (gfx1250) — matrix operand order repaired and checked by reading, not compiled, not run on hardware
//
#include <hip/hip_runtime.h>
#include <hip/hip_bf16.h>
#include <stdint.h>

// ---------------- problem constants ----------------
#define B_ 2
#define L_ 2048
#define D_ 512
#define H_ 8
#define HS_ 64
#define EPS_ 1e-5f
#define SCALE_ 0.125f   // HS^-0.5

typedef __bf16 bf16_t;
typedef __attribute__((ext_vector_type(16))) __bf16 v16bf;
typedef __attribute__((ext_vector_type(8)))  float  v8f;

union ABf {
    v16bf  v;
    uint4  q[2];
    bf16_t e[16];
};

static __device__ __forceinline__ v8f wmma_bf16(const ABf& a, const ABf& b, v8f c) {
    return __builtin_amdgcn_wmma_f32_16x16x32_bf16(
        /*neg_a=*/false, a.v, /*neg_b=*/false, b.v,
        /*c_mod=*/(short)0, c, /*reuse_a=*/false, /*reuse_b=*/false);
}

static __device__ __forceinline__ int imin(int a, int b) { return a < b ? a : b; }

// A-fragment (16x32, M on lanes[0..15], K in regs): per-lane halves load
// K = [16h .. 16h+15] contiguous -> two b128 loads from row `p`.
static __device__ __forceinline__ void load_afrag(ABf& a, const bf16_t* p, int kbase, int h) {
    a.q[0] = *(const uint4*)(p + kbase + 16 * h);
    a.q[1] = *(const uint4*)(p + kbase + 16 * h + 8);
}
// B-fragment (32x16, N on lanes, K in regs): per-lane 16 contiguous K starting
// at 16h within the fragment -> two b128 loads from row `p` (row = N index).
static __device__ __forceinline__ void load_bfrag(ABf& b, const bf16_t* p, int kbase, int h) {
    b.q[0] = *(const uint4*)(p + kbase + 16 * h);
    b.q[1] = *(const uint4*)(p + kbase + 16 * h + 8);
}

// ---------------- LayerNorm -> bf16 ----------------
__global__ void ln_kernel(const float* __restrict__ x, const float* __restrict__ g,
                          const float* __restrict__ bb, bf16_t* __restrict__ hn) {
    const int row = blockIdx.x;          // B*L rows
    const int tid = threadIdx.x;         // 256 threads
    const float* xr = x + (size_t)row * D_;
    float v0 = xr[tid], v1 = xr[tid + 256];
    float s = v0 + v1, ss = v0 * v0 + v1 * v1;
    #pragma unroll
    for (int m = 1; m < 32; m <<= 1) {
        s  += __shfl_xor(s,  m, 32);
        ss += __shfl_xor(ss, m, 32);
    }
    __shared__ float red[2][8];
    const int wid = tid >> 5;
    if ((tid & 31) == 0) { red[0][wid] = s; red[1][wid] = ss; }
    __syncthreads();
    float S = 0.f, SS = 0.f;
    #pragma unroll
    for (int i = 0; i < 8; ++i) { S += red[0][i]; SS += red[1][i]; }
    const float mu = S * (1.0f / D_);
    const float var = SS * (1.0f / D_) - mu * mu;
    const float rstd = rsqrtf(var + EPS_);
    bf16_t* hr = hn + (size_t)row * D_;
    hr[tid]       = (bf16_t)((v0 - mu) * rstd * g[tid]       + bb[tid]);
    hr[tid + 256] = (bf16_t)((v1 - mu) * rstd * g[tid + 256] + bb[tid + 256]);
}

// ---------------- f32 -> bf16 convert ----------------
__global__ void cvt_kernel(const float* __restrict__ s, bf16_t* __restrict__ d, int n) {
    int i = blockIdx.x * blockDim.x + threadIdx.x;
    if (i < n) d[i] = (bf16_t)s[i];
}

// ---------------- QKV projection GEMM (WMMA, 32x64 per-wave tile) ----------------
// C[r, j] = sum_k hn[r,k] * w[j,k] + bias[j]; scatter to [B,H,L,HS] bf16.
__global__ void gemm_qkv_kernel(const bf16_t* __restrict__ hn,
                                const bf16_t* __restrict__ wq, const bf16_t* __restrict__ wk,
                                const bf16_t* __restrict__ wv,
                                const float* __restrict__ bq, const float* __restrict__ bk,
                                const float* __restrict__ bv,
                                bf16_t* __restrict__ qo, bf16_t* __restrict__ ko,
                                bf16_t* __restrict__ vo) {
    const int z = blockIdx.z;
    const bf16_t* w    = (z == 0) ? wq : (z == 1) ? wk : wv;
    const float*  bias = (z == 0) ? bq : (z == 1) ? bk : bv;
    bf16_t*       dst  = (z == 0) ? qo : (z == 1) ? ko : vo;

    const int wt = blockIdx.x * 4 + (threadIdx.x >> 5);   // 1024 wave-tiles
    const int mt = wt >> 3, ng = wt & 7;
    const int m0 = mt * 32, n0 = ng * 64;
    const int t = threadIdx.x & 31, h = t >> 4, c = t & 15;

    const bf16_t* arow0 = hn + (size_t)(m0 + c) * D_;
    const bf16_t* arow1 = hn + (size_t)(m0 + 16 + c) * D_;

    v8f acc[2][4] = {{{}, {}, {}, {}}, {{}, {}, {}, {}}};
    #pragma unroll
    for (int k0 = 0; k0 < D_; k0 += 32) {
        ABf a0, a1;
        load_afrag(a0, arow0, k0, h);
        load_afrag(a1, arow1, k0, h);
        #pragma unroll
        for (int nb = 0; nb < 4; ++nb) {
            ABf b;
            load_bfrag(b, w + (size_t)(n0 + nb * 16 + c) * D_, k0, h);
            acc[0][nb] = wmma_bf16(a0, b, acc[0][nb]);
            acc[1][nb] = wmma_bf16(a1, b, acc[1][nb]);
        }
    }
    const int head = n0 >> 6;
    #pragma unroll
    for (int nb = 0; nb < 4; ++nb) {
        const int j  = n0 + nb * 16 + c;
        const int hs = j & 63;
        const float bj = bias[j];
        #pragma unroll
        for (int mi = 0; mi < 2; ++mi)
            #pragma unroll
            for (int v = 0; v < 8; ++v) {
                const int r = m0 + mi * 16 + v + 8 * h;      // r in [0, B*L)
                const int bb = r >> 11, l = r & (L_ - 1);
                dst[(((size_t)bb * H_ + head) * L_ + l) * HS_ + hs] = (bf16_t)(acc[mi][nb][v] + bj);
            }
    }
}

// ---------------- Flash attention with skewed relative bias ----------------
struct AttnSh {
    bf16_t p[16 * 32];   // P restriping tile (1 KB)
    bf16_t v[32 * 64];   // async-staged V chunk (4 KB)
};

__global__ void attn_kernel(const bf16_t* __restrict__ qm, const bf16_t* __restrict__ km,
                            const bf16_t* __restrict__ vm, const bf16_t* __restrict__ erm,
                            bf16_t* __restrict__ om) {
    __shared__ __align__(16) AttnSh sh[4];

    const int w   = threadIdx.x >> 5;
    const int l0  = (blockIdx.x * 4 + w) * 16;           // q-tile rows
    const int bh  = blockIdx.y;                          // b*H + head
    const int head = bh & (H_ - 1);
    const int t = threadIdx.x & 31, h = t >> 4, c = t & 15;

    const bf16_t* qb = qm  + (size_t)bh * L_ * HS_;
    const bf16_t* kb = km  + (size_t)bh * L_ * HS_;
    const bf16_t* vb = vm  + (size_t)bh * L_ * HS_;
    const bf16_t* eb = erm + (size_t)head * L_ * HS_;
    bf16_t* ps  = sh[w].p;
    bf16_t* vsh = sh[w].v;
    // LDS byte offset of this wave's V buffer (low 32 bits of flat address).
    const uint32_t vlds = (uint32_t)(size_t)(void*)vsh;

    // Q A-fragments over K(HS)=64
    ABf aq0, aq1;
    {
        const bf16_t* qr = qb + (size_t)(l0 + c) * HS_;
        load_afrag(aq0, qr, 0, h);
        load_afrag(aq1, qr, 32, h);
    }

    v8f acc[4] = {{}, {}, {}, {}};
    float mrow[8], lrow[8];
    #pragma unroll
    for (int v = 0; v < 8; ++v) { mrow[v] = -INFINITY; lrow[v] = 0.f; }

    for (int m0 = 0; m0 <= l0 + 15; m0 += 32) {
        // make sure prior DS reads of vsh retired before async writes land there
        asm volatile("s_wait_dscnt 0x0" ::: "memory");
        // ---- async-stage V[m0..m0+31][0..63] into LDS (16B per lane per op) ----
        #pragma unroll
        for (int it = 0; it < 8; ++it) {
            const int rloc = it * 4 + (t >> 3);
            const int row  = imin(m0 + rloc, L_ - 1);         // clamped rows are masked (p==0)
            const uint32_t lds_a = vlds + (uint32_t)(rloc * 128 + (t & 7) * 16);
            const bf16_t* g = vb + (size_t)row * HS_ + (t & 7) * 8;
            asm volatile("global_load_async_to_lds_b128 %0, %1, off"
                         :: "v"(lds_a), "v"(g) : "memory");
        }

        // prefetch next chunk's K rows while scores are computed
        __builtin_prefetch(kb + (size_t)imin(m0 + 32, L_ - 1) * HS_, 0, 1);

        float sc[2][8];
        #pragma unroll
        for (int sub = 0; sub < 2; ++sub) {
            const int ms = m0 + 16 * sub;
            // ---- q . k^T ----
            const int krow = imin(ms + c, L_ - 1);
            const bf16_t* kr = kb + (size_t)krow * HS_;
            ABf bk0, bk1;
            load_bfrag(bk0, kr, 0, h);
            load_bfrag(bk1, kr, 32, h);
            v8f qk = {};
            qk = wmma_bf16(aq0, bk0, qk);
            qk = wmma_bf16(aq1, bk1, qk);

            // ---- q . Er^T band: rows J..J+15 and J+16..J+31 ----
            const int J = (L_ - 16) - l0 + ms;           // >= 0
            const int e0 = imin(J + c,      L_ - 1);
            const int e1 = imin(J + 16 + c, L_ - 1);
            const bf16_t* er0 = eb + (size_t)e0 * HS_;
            const bf16_t* er1 = eb + (size_t)e1 * HS_;
            ABf be;
            v8f qe0 = {}, qe1 = {};
            load_bfrag(be, er0, 0, h);  qe0 = wmma_bf16(aq0, be, qe0);
            load_bfrag(be, er0, 32, h); qe0 = wmma_bf16(aq1, be, qe0);
            load_bfrag(be, er1, 0, h);  qe1 = wmma_bf16(aq0, be, qe1);
            load_bfrag(be, er1, 32, h); qe1 = wmma_bf16(aq1, be, qe1);

            // ---- skew gather + scale + causal mask ----
            // SRel(r,c) = QE[r, 15-r+c]: intra-half lane shuffle on C layout.
            #pragma unroll
            for (int v = 0; v < 8; ++v) {
                const int r = v + 8 * h;                 // local row
                const int src = 15 - r + c;              // 0..30
                const int lanesrc = h * 16 + (src & 15);
                const float s0 = __shfl(qe0[v], lanesrc, 32);
                const float s1 = __shfl(qe1[v], lanesrc, 32);
                const float srel = (src < 16) ? s0 : s1;
                const int grow = l0 + r, gcol = ms + c;
                const float val = qk[v] * SCALE_ + srel;
                sc[sub][v] = (gcol > grow) ? -INFINITY : val;
            }
        }

        // ---- online softmax over this 32-col chunk ----
        float alpha[8];
        #pragma unroll
        for (int v = 0; v < 8; ++v) {
            float rm = fmaxf(sc[0][v], sc[1][v]);
            #pragma unroll
            for (int m = 1; m < 16; m <<= 1) rm = fmaxf(rm, __shfl_xor(rm, m, 32));
            const float nm = fmaxf(mrow[v], rm);
            const float a = expf(mrow[v] - nm);
            const float p0 = expf(sc[0][v] - nm);
            const float p1 = expf(sc[1][v] - nm);
            sc[0][v] = p0; sc[1][v] = p1;
            float rs = p0 + p1;
            #pragma unroll
            for (int m = 1; m < 16; m <<= 1) rs += __shfl_xor(rs, m, 32);
            lrow[v] = lrow[v] * a + rs;
            mrow[v] = nm;
            alpha[v] = a;
        }
        #pragma unroll
        for (int nb = 0; nb < 4; ++nb)
            #pragma unroll
            for (int v = 0; v < 8; ++v) acc[nb][v] *= alpha[v];

        // ---- restripe P (C-layout -> A-layout) through LDS ----
        #pragma unroll
        for (int sub = 0; sub < 2; ++sub)
            #pragma unroll
            for (int v = 0; v < 8; ++v)
                ps[(v + 8 * h) * 32 + sub * 16 + c] = (bf16_t)sc[sub][v];
        asm volatile("s_wait_dscnt 0x0" ::: "memory");
        ABf ap;
        ap.q[0] = *(const uint4*)(ps + c * 32 + 16 * h);
        ap.q[1] = *(const uint4*)(ps + c * 32 + 16 * h + 8);

        // ---- wait for async V staging, then P @ V from LDS ----
        asm volatile("s_wait_asynccnt 0x0" ::: "memory");
        #pragma unroll
        for (int nb = 0; nb < 4; ++nb) {
            ABf bv;
            #pragma unroll
            for (int i = 0; i < 16; ++i)
                bv.e[i] = vsh[(16 * h + i) * HS_ + nb * 16 + c];
            acc[nb] = wmma_bf16(ap, bv, acc[nb]);
        }
    }

    // ---- finalize: O[b,l, head*64+hs] (bf16, [B*L, D] layout) ----
    #pragma unroll
    for (int nb = 0; nb < 4; ++nb)
        #pragma unroll
        for (int v = 0; v < 8; ++v) {
            const int r = l0 + v + 8 * h;
            const int bb = bh >> 3;
            const int j = head * HS_ + nb * 16 + c;
            om[((size_t)bb * L_ + r) * D_ + j] = (bf16_t)(acc[nb][v] / lrow[v]);
        }
}

// ---------------- output projection GEMM (WMMA, 32x64 tile, f32 out) ----------------
__global__ void gemm_out_kernel(const bf16_t* __restrict__ om, const bf16_t* __restrict__ wo,
                                const float* __restrict__ bo, float* __restrict__ out) {
    const int wt = blockIdx.x * 4 + (threadIdx.x >> 5);   // 1024 wave-tiles
    const int mt = wt >> 3, ng = wt & 7;
    const int m0 = mt * 32, n0 = ng * 64;
    const int t = threadIdx.x & 31, h = t >> 4, c = t & 15;

    const bf16_t* arow0 = om + (size_t)(m0 + c) * D_;
    const bf16_t* arow1 = om + (size_t)(m0 + 16 + c) * D_;

    v8f acc[2][4] = {{{}, {}, {}, {}}, {{}, {}, {}, {}}};
    #pragma unroll
    for (int k0 = 0; k0 < D_; k0 += 32) {
        ABf a0, a1;
        load_afrag(a0, arow0, k0, h);
        load_afrag(a1, arow1, k0, h);
        #pragma unroll
        for (int nb = 0; nb < 4; ++nb) {
            ABf b;
            load_bfrag(b, wo + (size_t)(n0 + nb * 16 + c) * D_, k0, h);
            acc[0][nb] = wmma_bf16(a0, b, acc[0][nb]);
            acc[1][nb] = wmma_bf16(a1, b, acc[1][nb]);
        }
    }
    #pragma unroll
    for (int nb = 0; nb < 4; ++nb) {
        const int j = n0 + nb * 16 + c;
        const float bj = bo[j];
        #pragma unroll
        for (int mi = 0; mi < 2; ++mi)
            #pragma unroll
            for (int v = 0; v < 8; ++v) {
                const int r = m0 + mi * 16 + v + 8 * h;
                out[(size_t)r * D_ + j] = acc[mi][nb][v] + bj;
            }
    }
}

// ---------------- host launch ----------------
extern "C" void kernel_launch(void* const* d_in, const int* in_sizes, int n_in,
                              void* d_out, int out_size, void* d_ws, size_t ws_size,
                              hipStream_t stream) {
    const float* x    = (const float*)d_in[0];
    // d_in[1] = mask (bool) -- causal mask applied analytically
    const float* ln_g = (const float*)d_in[2];
    const float* ln_b = (const float*)d_in[3];
    const float* wq   = (const float*)d_in[4];
    const float* bq   = (const float*)d_in[5];
    const float* wk   = (const float*)d_in[6];
    const float* bk   = (const float*)d_in[7];
    const float* wv   = (const float*)d_in[8];
    const float* bv   = (const float*)d_in[9];
    const float* wo   = (const float*)d_in[10];
    const float* bo   = (const float*)d_in[11];
    const float* Er   = (const float*)d_in[12];
    float* out = (float*)d_out;

    char* ws = (char*)d_ws;
    size_t off = 0;
    auto carve = [&](size_t bytes) { char* p = ws + off; off += (bytes + 255) & ~(size_t)255; return p; };
    bf16_t* hnbf = (bf16_t*)carve((size_t)B_ * L_ * D_ * 2);
    bf16_t* wqbf = (bf16_t*)carve((size_t)D_ * D_ * 2);
    bf16_t* wkbf = (bf16_t*)carve((size_t)D_ * D_ * 2);
    bf16_t* wvbf = (bf16_t*)carve((size_t)D_ * D_ * 2);
    bf16_t* wobf = (bf16_t*)carve((size_t)D_ * D_ * 2);
    bf16_t* erbf = (bf16_t*)carve((size_t)H_ * L_ * HS_ * 2);
    bf16_t* qbf  = (bf16_t*)carve((size_t)B_ * H_ * L_ * HS_ * 2);
    bf16_t* kbf  = (bf16_t*)carve((size_t)B_ * H_ * L_ * HS_ * 2);
    bf16_t* vbf  = (bf16_t*)carve((size_t)B_ * H_ * L_ * HS_ * 2);
    bf16_t* obf  = (bf16_t*)carve((size_t)B_ * L_ * D_ * 2);

    // 1) LayerNorm -> bf16
    ln_kernel<<<B_ * L_, 256, 0, stream>>>(x, ln_g, ln_b, hnbf);

    // 2) weight / Er converts
    const int nw = D_ * D_;
    cvt_kernel<<<(nw + 255) / 256, 256, 0, stream>>>(wq, wqbf, nw);
    cvt_kernel<<<(nw + 255) / 256, 256, 0, stream>>>(wk, wkbf, nw);
    cvt_kernel<<<(nw + 255) / 256, 256, 0, stream>>>(wv, wvbf, nw);
    cvt_kernel<<<(nw + 255) / 256, 256, 0, stream>>>(wo, wobf, nw);
    const int ne = H_ * L_ * HS_;
    cvt_kernel<<<(ne + 255) / 256, 256, 0, stream>>>(Er, erbf, ne);

    // 3) Q/K/V projections: 1024 wave-tiles (32x64) per matrix, z selects q/k/v
    gemm_qkv_kernel<<<dim3(256, 1, 3), 128, 0, stream>>>(
        hnbf, wqbf, wkbf, wvbf, bq, bk, bv, qbf, kbf, vbf);

    // 4) skewed-relative flash attention: 4 waves/block, 16 q-rows per wave
    attn_kernel<<<dim3(L_ / 64, B_ * H_), 128, 0, stream>>>(qbf, kbf, vbf, erbf, obf);

    // 5) output projection -> f32 d_out
    gemm_out_kernel<<<256, 128, 0, stream>>>(obf, wobf, bo, out);
}
